// Sequence_34789235098039
// MI455X (gfx1250) — matrix operand order repaired and checked by reading, not compiled, not run on hardware
//
#include <hip/hip_runtime.h>

#define D_IN  512
#define HID   1024
#define D_OUT 512
#define TSEQ  256
#define BATCH 64
#define FUT   32
#define TTOT  (TSEQ + FUT)

typedef __attribute__((ext_vector_type(16))) __bf16 v16bf;
typedef __attribute__((ext_vector_type(8)))  float  v8f;

union Frag16 { v16bf v; float4 f[2]; };

__device__ __forceinline__ unsigned short f2bf(float x) {
    unsigned int u = __float_as_uint(x);
    u += 0x7fffu + ((u >> 16) & 1u);          // round-to-nearest-even
    return (unsigned short)(u >> 16);
}

__device__ __forceinline__ float sigf(float x) {
    return 1.f / (1.f + __expf(-x));
}

// A fragment: 16x32 bf16, row-major activations [rows][ld].
// Per ISA: lanes 0-15 hold M=lane, VGPR0-3 = K 0..7, VGPR4-7 = K 16..23;
//          lanes 16-31 hold M=lane-16, K 8..15 and K 24..31.
__device__ __forceinline__ v16bf load_a(const unsigned short* __restrict__ base,
                                        int ld, int row0, int k0, int lane) {
    const int r  = row0 + (lane & 15);
    const int hs = lane >> 4;
    const char* p = (const char*)(base + (size_t)r * ld + k0);
    Frag16 u;
    u.f[0] = *(const float4*)(p + hs * 16);        // K = k0 + hs*8 .. +8
    u.f[1] = *(const float4*)(p + 32 + hs * 16);   // K = k0+16+hs*8 .. +8
    return u.v;
}

// B fragment: 32x16 bf16 (K x N). Weights stored [N][K] row-major (torch layout).
// Lane l holds column N = n0 + (l&15), K = k0 + (l>>4)*8 .. +8 and k0+16+(l>>4)*8 .. +8.
__device__ __forceinline__ v16bf load_b(const unsigned short* __restrict__ W,
                                        int ld, int n0, int k0, int lane) {
    const float4* p = (const float4*)(W + (size_t)(n0 + (lane & 15)) * ld
                                        + k0 + (lane >> 4) * 8);
    Frag16 u;
    u.f[0] = p[0];
    u.f[1] = p[2];
    return u.v;
}

__device__ __forceinline__ v8f wmma_bf16(v16bf a, v16bf b, v8f c) {
    return __builtin_amdgcn_wmma_f32_16x16x32_bf16(false, a, false, b,
                                                   (short)0, c, false, false);
}

// ---------------------------------------------------------------------------
// Layer 1: gates = x @ W_ih1^T + h1 @ W_hh1^T + bias; fused LSTM cell.
// Grid = HID/16 column tiles; block = 256 threads (8 waves).
// Waves 0-3: row tile rt = wave, K-half 0.  Waves 4-7: same rt, K-half 1.
// Split-K reduced through LDS, epilogue by waves 0-3.
// ---------------------------------------------------------------------------
__global__ void __launch_bounds__(256)
lstm_l1_step(const unsigned short* __restrict__ x,     // [64][512] bf16
             const unsigned short* __restrict__ Wih,   // [4096][512] bf16
             const unsigned short* __restrict__ Whh,   // [4096][1024] bf16
             const float* __restrict__ bias,           // [4096] (b_ih + b_hh)
             const unsigned short* __restrict__ h_in,  // [64][1024] bf16
             unsigned short* __restrict__ h_out,       // [64][1024] bf16
             float* __restrict__ c)                    // [64][1024] f32
{
    __shared__ float red[4 * 4 * 32 * 8];              // 16 KB
    const int lane = threadIdx.x & 31;
    const int wave = threadIdx.x >> 5;
    const int rt   = wave & 3;
    const int kh   = wave >> 2;
    const int ct   = blockIdx.x;                       // hidden tile [0,64)
    const int row0 = rt * 16;
    const int n0   = ct * 16;

    v8f acc[4] = {};                                   // i, f, g, o tiles

    if (kh == 0) {
        // x contribution (K = 512) + first quarter of h (K = 0..256): 24 iters
        for (int k0 = 0; k0 < D_IN; k0 += 32) {
            v16bf a = load_a(x, D_IN, row0, k0, lane);
#pragma unroll
            for (int q = 0; q < 4; ++q)
                acc[q] = wmma_bf16(a, load_b(Wih, D_IN, q * HID + n0, k0, lane), acc[q]);
        }
        for (int k0 = 0; k0 < 256; k0 += 32) {
            v16bf a = load_a(h_in, HID, row0, k0, lane);
#pragma unroll
            for (int q = 0; q < 4; ++q)
                acc[q] = wmma_bf16(a, load_b(Whh, HID, q * HID + n0, k0, lane), acc[q]);
        }
    } else {
        // remaining h (K = 256..1024): 24 iters
        for (int k0 = 256; k0 < HID; k0 += 32) {
            v16bf a = load_a(h_in, HID, row0, k0, lane);
#pragma unroll
            for (int q = 0; q < 4; ++q)
                acc[q] = wmma_bf16(a, load_b(Whh, HID, q * HID + n0, k0, lane), acc[q]);
        }
        float* dst = &red[(((size_t)rt * 4) * 32 + lane) * 8];
#pragma unroll
        for (int q = 0; q < 4; ++q)
#pragma unroll
            for (int v = 0; v < 8; ++v)
                dst[(size_t)q * 32 * 8 + v] = acc[q][v];
    }
    __syncthreads();

    if (kh == 0) {
        const float* src = &red[(((size_t)rt * 4) * 32 + lane) * 8];
#pragma unroll
        for (int q = 0; q < 4; ++q)
#pragma unroll
            for (int v = 0; v < 8; ++v)
                acc[q][v] += src[(size_t)q * 32 * 8 + v];

        // C/D layout: lane l, vgpr v -> row = v + 8*(l>>4), col = l&15
        const int colg = n0 + (lane & 15);
        const int hs   = lane >> 4;
        const float bi = bias[0 * HID + colg];
        const float bf_ = bias[1 * HID + colg];
        const float bg = bias[2 * HID + colg];
        const float bo = bias[3 * HID + colg];
#pragma unroll
        for (int v = 0; v < 8; ++v) {
            const int row = row0 + hs * 8 + v;
            const float ig = sigf(acc[0][v] + bi);
            const float fg = sigf(acc[1][v] + bf_);
            const float gg = tanhf(acc[2][v] + bg);
            const float og = sigf(acc[3][v] + bo);
            const size_t idx = (size_t)row * HID + colg;
            const float cn = fg * c[idx] + ig * gg;
            c[idx] = cn;
            h_out[idx] = f2bf(og * tanhf(cn));
        }
    }
}

// ---------------------------------------------------------------------------
// Layer 2: gates = h1 @ W_ih2^T + h2 @ W_hh2^T + bias; fused cell + output.
// Grid = D_OUT/16 tiles; same split-K wave layout.
// ---------------------------------------------------------------------------
__global__ void __launch_bounds__(256)
lstm_l2_step(const unsigned short* __restrict__ h1,    // [64][1024] bf16
             const unsigned short* __restrict__ Wih,   // [2048][1024] bf16
             const unsigned short* __restrict__ Whh,   // [2048][512] bf16
             const float* __restrict__ bias,           // [2048]
             const unsigned short* __restrict__ h_in,  // [64][512] bf16
             unsigned short* __restrict__ h_out,       // [64][512] bf16
             float* __restrict__ c,                    // [64][512] f32
             float* __restrict__ out,                  // [64][288][512] f32
             int t)
{
    __shared__ float red[4 * 4 * 32 * 8];
    const int lane = threadIdx.x & 31;
    const int wave = threadIdx.x >> 5;
    const int rt   = wave & 3;
    const int kh   = wave >> 2;
    const int ct   = blockIdx.x;                       // out tile [0,32)
    const int row0 = rt * 16;
    const int n0   = ct * 16;

    v8f acc[4] = {};

    if (kh == 0) {
        // h1 K = 0..768 : 24 iters
        for (int k0 = 0; k0 < 768; k0 += 32) {
            v16bf a = load_a(h1, HID, row0, k0, lane);
#pragma unroll
            for (int q = 0; q < 4; ++q)
                acc[q] = wmma_bf16(a, load_b(Wih, HID, q * D_OUT + n0, k0, lane), acc[q]);
        }
    } else {
        // h1 K = 768..1024 (8 iters) + h2 K = 0..512 (16 iters)
        for (int k0 = 768; k0 < HID; k0 += 32) {
            v16bf a = load_a(h1, HID, row0, k0, lane);
#pragma unroll
            for (int q = 0; q < 4; ++q)
                acc[q] = wmma_bf16(a, load_b(Wih, HID, q * D_OUT + n0, k0, lane), acc[q]);
        }
        for (int k0 = 0; k0 < D_OUT; k0 += 32) {
            v16bf a = load_a(h_in, D_OUT, row0, k0, lane);
#pragma unroll
            for (int q = 0; q < 4; ++q)
                acc[q] = wmma_bf16(a, load_b(Whh, D_OUT, q * D_OUT + n0, k0, lane), acc[q]);
        }
        float* dst = &red[(((size_t)rt * 4) * 32 + lane) * 8];
#pragma unroll
        for (int q = 0; q < 4; ++q)
#pragma unroll
            for (int v = 0; v < 8; ++v)
                dst[(size_t)q * 32 * 8 + v] = acc[q][v];
    }
    __syncthreads();

    if (kh == 0) {
        const float* src = &red[(((size_t)rt * 4) * 32 + lane) * 8];
#pragma unroll
        for (int q = 0; q < 4; ++q)
#pragma unroll
            for (int v = 0; v < 8; ++v)
                acc[q][v] += src[(size_t)q * 32 * 8 + v];

        const int colg = n0 + (lane & 15);
        const int hs   = lane >> 4;
        const float bi = bias[0 * D_OUT + colg];
        const float bf_ = bias[1 * D_OUT + colg];
        const float bg = bias[2 * D_OUT + colg];
        const float bo = bias[3 * D_OUT + colg];
#pragma unroll
        for (int v = 0; v < 8; ++v) {
            const int row = row0 + hs * 8 + v;
            const float ig = sigf(acc[0][v] + bi);
            const float fg = sigf(acc[1][v] + bf_);
            const float gg = tanhf(acc[2][v] + bg);
            const float og = sigf(acc[3][v] + bo);
            const size_t idx = (size_t)row * D_OUT + colg;
            const float cn = fg * c[idx] + ig * gg;
            c[idx] = cn;
            const float hv = og * tanhf(cn);
            h_out[idx] = f2bf(hv);
            out[(size_t)row * (TTOT * D_OUT) + (size_t)t * D_OUT + colg] = hv;
        }
    }
}

// ------------------------- setup kernels -----------------------------------
__global__ void k_cvt_bf16(const float* __restrict__ s, unsigned short* __restrict__ d, int n) {
    int i = blockIdx.x * blockDim.x + threadIdx.x;
    if (i < n) d[i] = f2bf(s[i]);
}
__global__ void k_add_f32(const float* __restrict__ a, const float* __restrict__ b,
                          float* __restrict__ d, int n) {
    int i = blockIdx.x * blockDim.x + threadIdx.x;
    if (i < n) d[i] = a[i] + b[i];
}
__global__ void k_zero_f32(float* __restrict__ p, int n) {
    int i = blockIdx.x * blockDim.x + threadIdx.x;
    if (i < n) p[i] = 0.f;
}

// ---------------------------------------------------------------------------
extern "C" void kernel_launch(void* const* d_in, const int* in_sizes, int n_in,
                              void* d_out, int out_size, void* d_ws, size_t ws_size,
                              hipStream_t stream) {
    const float* input = (const float*)d_in[0];
    const float* W_ih1 = (const float*)d_in[1];
    const float* W_hh1 = (const float*)d_in[2];
    const float* b_ih1 = (const float*)d_in[3];
    const float* b_hh1 = (const float*)d_in[4];
    const float* W_ih2 = (const float*)d_in[5];
    const float* W_hh2 = (const float*)d_in[6];
    const float* b_ih2 = (const float*)d_in[7];
    const float* b_hh2 = (const float*)d_in[8];
    float* out = (float*)d_out;

    // ------------- workspace carve (all chunks 256B-aligned sizes) ---------
    char* p = (char*)d_ws;
    unsigned short* wih1 = (unsigned short*)p; p += (size_t)4 * HID * D_IN * 2;     // 4 MB
    unsigned short* whh1 = (unsigned short*)p; p += (size_t)4 * HID * HID * 2;      // 8 MB
    unsigned short* wih2 = (unsigned short*)p; p += (size_t)4 * D_OUT * HID * 2;    // 4 MB
    unsigned short* whh2 = (unsigned short*)p; p += (size_t)4 * D_OUT * D_OUT * 2;  // 2 MB
    unsigned short* xbf  = (unsigned short*)p; p += (size_t)TSEQ * BATCH * D_IN * 2;// 16 MB
    float* b1 = (float*)p; p += (size_t)4 * HID * 4;
    float* b2 = (float*)p; p += (size_t)4 * D_OUT * 4;
    char* state = p;
    unsigned short* h1b[2];
    h1b[0] = (unsigned short*)p; p += (size_t)BATCH * HID * 2;
    h1b[1] = (unsigned short*)p; p += (size_t)BATCH * HID * 2;
    unsigned short* h2b[2];
    h2b[0] = (unsigned short*)p; p += (size_t)BATCH * D_OUT * 2;
    h2b[1] = (unsigned short*)p; p += (size_t)BATCH * D_OUT * 2;
    float* c1 = (float*)p; p += (size_t)BATCH * HID * 4;
    float* c2 = (float*)p; p += (size_t)BATCH * D_OUT * 4;
    const int state_floats = (int)((p - state) / 4);

    // ------------- one-time (per launch) precision conversion --------------
    const int TPB = 256;
    auto grid = [](int n) { return dim3((n + 255) / 256); };
    k_cvt_bf16<<<grid(4 * HID * D_IN), TPB, 0, stream>>>(W_ih1, wih1, 4 * HID * D_IN);
    k_cvt_bf16<<<grid(4 * HID * HID), TPB, 0, stream>>>(W_hh1, whh1, 4 * HID * HID);
    k_cvt_bf16<<<grid(4 * D_OUT * HID), TPB, 0, stream>>>(W_ih2, wih2, 4 * D_OUT * HID);
    k_cvt_bf16<<<grid(4 * D_OUT * D_OUT), TPB, 0, stream>>>(W_hh2, whh2, 4 * D_OUT * D_OUT);
    k_cvt_bf16<<<grid(TSEQ * BATCH * D_IN), TPB, 0, stream>>>(input, xbf, TSEQ * BATCH * D_IN);
    k_add_f32<<<grid(4 * HID), TPB, 0, stream>>>(b_ih1, b_hh1, b1, 4 * HID);
    k_add_f32<<<grid(4 * D_OUT), TPB, 0, stream>>>(b_ih2, b_hh2, b2, 4 * D_OUT);
    k_zero_f32<<<grid(state_floats), TPB, 0, stream>>>((float*)state, state_floats);

    // ------------- recurrence: 2 kernels per step, stream-ordered ----------
    for (int t = 0; t < TTOT; ++t) {
        const unsigned short* x =
            (t < TSEQ) ? (xbf + (size_t)t * BATCH * D_IN) : h2b[t & 1];
        lstm_l1_step<<<HID / 16, 256, 0, stream>>>(
            x, wih1, whh1, b1, h1b[t & 1], h1b[(t + 1) & 1], c1);
        lstm_l2_step<<<D_OUT / 16, 256, 0, stream>>>(
            h1b[(t + 1) & 1], wih2, whh2, b2, h2b[t & 1], h2b[(t + 1) & 1], c2, out, t);
    }
}
